// spatial_mamba_2181843386674
// MI455X (gfx1250) — compile-verified, matrix operand order checked by reading, not run on hardware
//
#include <hip/hip_runtime.h>
#include <hip/hip_bf16.h>
#include <math.h>

// ---------------------------------------------------------------------------
// Problem constants (match reference)
// ---------------------------------------------------------------------------
constexpr int B_  = 128;
constexpr int L_  = 31;
constexpr int D_  = 1024;
constexpr int DI_ = 2048;
constexpr int DS_ = 16;
constexpr int DC_ = 5;
constexpr int DTR_ = 64;
constexpr int XD_ = DTR_ + 2 * DS_;      // 96  (x_dbl width)

typedef __attribute__((ext_vector_type(16))) _Float16 v16h;
typedef __attribute__((ext_vector_type(8)))  float    v8f;
typedef __attribute__((ext_vector_type(4)))  float    v4f;

// Global-address-space pointer types: force global_load/global_store codegen.
typedef const __attribute__((address_space(1))) v4f*   gv4f_cp;
typedef const __attribute__((address_space(1))) float* gf_cp;
typedef __attribute__((address_space(1))) float*       gf_p;

// ---------------------------------------------------------------------------
// Helpers
// ---------------------------------------------------------------------------
__device__ __forceinline__ v16h frag_from_f32(const float* p) {
  // 16 consecutive f32 -> 16 packed f16 (one lane's WMMA operand)
  gv4f_cp g = (gv4f_cp)p;
  v16h r;
#pragma unroll
  for (int i = 0; i < 4; ++i) {
    v4f f = g[i];
    r[4 * i + 0] = (_Float16)f.x;
    r[4 * i + 1] = (_Float16)f.y;
    r[4 * i + 2] = (_Float16)f.z;
    r[4 * i + 3] = (_Float16)f.w;
  }
  return r;
}

__device__ __forceinline__ float silu_f(float v) {
  return v / (1.0f + __expf(-v));
}

__device__ __forceinline__ float act_apply(float v, int act) {
  if (act == 1) return silu_f(v);                                // SiLU
  if (act == 2) return (v > 20.0f) ? v : log1pf(__expf(v));      // softplus
  return v;
}

// ---------------------------------------------------------------------------
// WMMA GEMM:  C[m,n] = act( sum_k A[m,k]*W[n,k] + bias[n] + addend[m,n] )
// A: (M,K) row-major lda ; W: (N,K) row-major ldw ; C/addend: (M,N) ldc=N
// One wave computes a 32(M) x (16*JT)(N) tile:
//   2 A-fragments x JT B-fragments -> 2*JT v_wmma per 32-deep k-chunk.
// JT is chosen so N % (16*JT) == 0 at every call site: the k-loop is fully
// branch-free straight-line code (no tile-validity phis, no accumulator
// copies, EXEC all-ones through every WMMA as the ISA requires).
// Requires M % 32 == 0, K % 32 == 0, N % (16*JT) == 0.
// ---------------------------------------------------------------------------
template <int JT>
__global__ void __launch_bounds__(256)
wmma_gemm_kernel(const float* __restrict__ A, int lda,
                 const float* __restrict__ W, int ldw,
                 const float* __restrict__ bias,
                 const float* __restrict__ addend,
                 float* __restrict__ C, int ldc,
                 int M, int N, int K, int act)
{
  // Scalar (SGPR) wave id: readfirstlane(threadIdx.x) is lane 0's tid.
  const int wib  = __builtin_amdgcn_readfirstlane((int)threadIdx.x) >> 5;
  const int wave = (int)blockIdx.x * 8 + wib;         // 256 threads = 8 waves
  const int lane = threadIdx.x & 31;

  const int tilesN = N / (16 * JT);            // exact by construction
  const int tm     = wave / tilesN;
  const int tn     = wave % tilesN;
  const int m0     = tm << 5;                  // 32 rows per wave tile
  const int n0     = tn * (16 * JT);           // 16*JT cols per wave tile
  if (m0 >= M) return;

  const int lhalf = lane & 15;                 // row (A) / col (B) within tile
  const int khalf = (lane >> 4) << 4;          // lanes 16..31 own K+16..K+31

  v8f acc0[JT] = {};                           // rows m0 .. m0+15
  v8f acc1[JT] = {};                           // rows m0+16 .. m0+31

  const float* ap0 = A + (size_t)(m0 + lhalf) * lda + khalf;
  const float* ap1 = ap0 + (size_t)16 * lda;
  const float* wp[JT];
#pragma unroll
  for (int j = 0; j < JT; ++j)
    wp[j] = W + (size_t)(n0 + (j << 4) + lhalf) * ldw + khalf;

  for (int k0 = 0; k0 < K; k0 += 32) {
    // Speculative prefetch 2 chunks ahead (OOB speculative prefetch is
    // silently dropped per ISA) -> global_prefetch_b8, no guard branch.
    __builtin_prefetch(ap0 + 64, 0, 1);
    __builtin_prefetch(ap1 + 64, 0, 1);

    const v16h a0 = frag_from_f32(ap0);
    const v16h a1 = frag_from_f32(ap1);
#pragma unroll
    for (int j = 0; j < JT; ++j) {
      const v16h bf = frag_from_f32(wp[j]);
      acc0[j] = __builtin_amdgcn_wmma_f32_16x16x32_f16(
          false, a0, false, bf, (short)0, acc0[j], false, false);
      acc1[j] = __builtin_amdgcn_wmma_f32_16x16x32_f16(
          false, a1, false, bf, (short)0, acc1[j], false, false);
      wp[j] += 32;
    }
    ap0 += 32;
    ap1 += 32;
  }

  // C/D layout: lane l, vgpr r -> row = r + 8*(l>=16), col = (l&15)
  const int rb = m0 + ((lane >> 4) << 3);
  gf_p  Cg = (gf_p)C;
  gf_cp Ag = (gf_cp)addend;
#pragma unroll
  for (int j = 0; j < JT; ++j) {
    const int col = n0 + (j << 4) + lhalf;
    const float bv = bias ? bias[col] : 0.0f;
#pragma unroll
    for (int r = 0; r < 8; ++r) {
      {
        const int row = rb + r;
        float v = acc0[j][r] + bv;
        if (addend) v += Ag[(size_t)row * ldc + col];
        Cg[(size_t)row * ldc + col] = act_apply(v, act);
      }
      {
        const int row = rb + 16 + r;
        float v = acc1[j][r] + bv;
        if (addend) v += Ag[(size_t)row * ldc + col];
        Cg[(size_t)row * ldc + col] = act_apply(v, act);
      }
    }
  }
}

// ---------------------------------------------------------------------------
// LayerNorm over last dim (D=1024), one 256-thread block per row (8 waves).
// ---------------------------------------------------------------------------
__global__ void __launch_bounds__(256)
layernorm_kernel(const float* __restrict__ X, const float* __restrict__ w,
                 const float* __restrict__ b, float* __restrict__ Y)
{
  const int row = blockIdx.x;
  const int t   = threadIdx.x;
  gv4f_cp x = (gv4f_cp)(X + (size_t)row * D_);

  v4f v = x[t];                                       // 256*4 == 1024
  float s  = v.x + v.y + v.z + v.w;
  float sq = v.x * v.x + v.y * v.y + v.z * v.z + v.w * v.w;

#pragma unroll
  for (int o = 16; o > 0; o >>= 1) {
    s  += __shfl_xor(s, o, 32);
    sq += __shfl_xor(sq, o, 32);
  }
  __shared__ float ss[8], ssq[8];
  if ((t & 31) == 0) { ss[t >> 5] = s; ssq[t >> 5] = sq; }
  __syncthreads();

  float tot = 0.0f, totq = 0.0f;
#pragma unroll
  for (int i = 0; i < 8; ++i) { tot += ss[i]; totq += ssq[i]; }

  const float mean = tot * (1.0f / D_);
  const float var  = totq * (1.0f / D_) - mean * mean;
  const float rs   = rsqrtf(var + 1e-5f);

  v4f wv = ((gv4f_cp)w)[t];
  v4f bv = ((gv4f_cp)b)[t];
  v4f o;
  o.x = (v.x - mean) * rs * wv.x + bv.x;
  o.y = (v.y - mean) * rs * wv.y + bv.y;
  o.z = (v.z - mean) * rs * wv.z + bv.z;
  o.w = (v.w - mean) * rs * wv.w + bv.w;
  ((__attribute__((address_space(1))) v4f*)(Y + (size_t)row * D_))[t] = o;
}

// ---------------------------------------------------------------------------
// Causal depthwise conv1d (kernel DC_) + SiLU.
// u lives in the first DI_ columns of xz (ld = 2*DI_); result -> uc (ld = DI_)
// ---------------------------------------------------------------------------
__global__ void __launch_bounds__(256)
conv_silu_kernel(const float* __restrict__ xz, const float* __restrict__ cw,
                 const float* __restrict__ cb, float* __restrict__ uc)
{
  const int idx = (int)(blockIdx.x * blockDim.x + threadIdx.x);
  if (idx >= B_ * L_ * DI_) return;
  const int d = idx % DI_;
  const int l = (idx / DI_) % L_;
  const int b = idx / (DI_ * L_);

  float s = cb[d];
#pragma unroll
  for (int k = 0; k < DC_; ++k) {
    const int lt = l + k - (DC_ - 1);
    if (lt >= 0)
      s += xz[(size_t)(b * L_ + lt) * (2 * DI_) + d] * cw[d * DC_ + k];
  }
  uc[idx] = silu_f(s);
}

// ---------------------------------------------------------------------------
// Selective-scan (L=31, DS=16 states in registers) fused with D*u skip and
// SiLU(z) gate.  One thread per (b, d).  Reads u from u_y, writes gated y
// back in-place (element read-then-written per step within one thread).
// ---------------------------------------------------------------------------
__global__ void __launch_bounds__(256)
ssm_scan_kernel(const float* __restrict__ dt,     // (B,L,DI)
                const float* __restrict__ xdbl,   // (B,L,96): Bm@64, Cm@80
                const float* __restrict__ xz,     // (B,L,4096): z at col DI_
                const float* __restrict__ A_log,  // (DI,DS)
                const float* __restrict__ Dssm,   // (DI)
                float* __restrict__ u_y)          // in: u, out: gated y
{
  const int idx = (int)(blockIdx.x * blockDim.x + threadIdx.x);
  if (idx >= B_ * DI_) return;
  const int d = idx % DI_;
  const int b = idx / DI_;

  float Ad[DS_];
#pragma unroll
  for (int s = 0; s < DS_; ++s) Ad[s] = -__expf(A_log[d * DS_ + s]);
  const float Dv = Dssm[d];

  float h[DS_];
#pragma unroll
  for (int s = 0; s < DS_; ++s) h[s] = 0.0f;

  for (int t = 0; t < L_; ++t) {
    const size_t rl = (size_t)(b * L_ + t);
    const float dtv = dt[rl * DI_ + d];
    const float uv  = u_y[rl * DI_ + d];
    const float* xd = xdbl + rl * XD_;
    float y = 0.0f;
#pragma unroll
    for (int s = 0; s < DS_; ++s) {
      h[s] = h[s] * __expf(dtv * Ad[s]) + dtv * xd[DTR_ + s] * uv;
      y += h[s] * xd[DTR_ + DS_ + s];
    }
    const float zv = xz[rl * (2 * DI_) + DI_ + d];
    u_y[rl * DI_ + d] = (y + uv * Dv) * silu_f(zv);
  }
}

// ---------------------------------------------------------------------------
// Classifier head stage 2: out[b] = c[b,:] . cls_W2[0,:] + cls_b2
// One 256-thread block per batch row.
// ---------------------------------------------------------------------------
__global__ void __launch_bounds__(256)
cls_head_kernel(const float* __restrict__ c, const float* __restrict__ w2,
                const float* __restrict__ b2, float* __restrict__ out)
{
  const int b = blockIdx.x;
  const int t = threadIdx.x;
  v4f cv = ((gv4f_cp)(c + (size_t)b * D_))[t];
  v4f wv = ((gv4f_cp)w2)[t];
  float s = cv.x * wv.x + cv.y * wv.y + cv.z * wv.z + cv.w * wv.w;
#pragma unroll
  for (int o = 16; o > 0; o >>= 1) s += __shfl_xor(s, o, 32);
  __shared__ float ps[8];
  if ((t & 31) == 0) ps[t >> 5] = s;
  __syncthreads();
  if (t == 0) {
    float tot = 0.0f;
#pragma unroll
    for (int i = 0; i < 8; ++i) tot += ps[i];
    out[b] = tot + b2[0];
  }
}

// ---------------------------------------------------------------------------
// Host-side launcher
// ---------------------------------------------------------------------------
static inline void launch_gemm(const float* A, int lda, const float* W, int ldw,
                               const float* bias, const float* add,
                               float* C, int ldc, int M, int N, int K, int act,
                               hipStream_t stream)
{
  const long tilesM = (M + 31) / 32;           // 32 rows per wave tile
  if (N % 64 == 0) {
    const long tilesN = N / 64;
    const long blocks = (tilesM * tilesN * 32 + 255) / 256;
    wmma_gemm_kernel<4><<<dim3((unsigned)blocks), dim3(256), 0, stream>>>(
        A, lda, W, ldw, bias, add, C, ldc, M, N, K, act);
  } else {                                     // N % 32 == 0 (x_proj, N=96)
    const long tilesN = N / 32;
    const long blocks = (tilesM * tilesN * 32 + 255) / 256;
    wmma_gemm_kernel<2><<<dim3((unsigned)blocks), dim3(256), 0, stream>>>(
        A, lda, W, ldw, bias, add, C, ldc, M, N, K, act);
  }
}

extern "C" void kernel_launch(void* const* d_in, const int* in_sizes, int n_in,
                              void* d_out, int out_size, void* d_ws, size_t ws_size,
                              hipStream_t stream)
{
  (void)in_sizes; (void)n_in; (void)out_size; (void)ws_size;

  const float* x         = (const float*)d_in[0];
  const float* norm1_w   = (const float*)d_in[1];
  const float* norm1_b   = (const float*)d_in[2];
  const float* in_proj_W = (const float*)d_in[3];
  const float* conv_w    = (const float*)d_in[4];
  const float* conv_b    = (const float*)d_in[5];
  const float* x_proj_W  = (const float*)d_in[6];
  const float* dt_proj_W = (const float*)d_in[7];
  const float* dt_proj_b = (const float*)d_in[8];
  const float* A_log     = (const float*)d_in[9];
  const float* D_ssm     = (const float*)d_in[10];
  const float* out_projW = (const float*)d_in[11];
  const float* norm2_w   = (const float*)d_in[12];
  const float* norm2_b   = (const float*)d_in[13];
  const float* ffn_W1    = (const float*)d_in[14];
  const float* ffn_b1    = (const float*)d_in[15];
  const float* ffn_W2    = (const float*)d_in[16];
  const float* ffn_b2    = (const float*)d_in[17];
  const float* cls_W1    = (const float*)d_in[18];
  const float* cls_b1    = (const float*)d_in[19];
  const float* cls_W2    = (const float*)d_in[20];
  const float* cls_b2    = (const float*)d_in[21];

  constexpr size_t BL = (size_t)B_ * L_;       // 3968 rows
  const int M = (int)BL;

  // Workspace layout (floats)
  float* ws   = (float*)d_ws;
  float* hln  = ws;                            // (BL, D)      - LN scratch (reused)
  float* xz   = hln  + BL * D_;                // (BL, 2*DI)   - u | z
  float* uc   = xz   + BL * 2 * DI_;           // (BL, DI)     - conv out, then gated y
  float* xdbl = uc   + BL * DI_;               // (BL, 96)     - dt|B|C
  float* dtb  = xdbl + BL * XD_;               // (BL, DI)     - softplus(dt_proj)
  float* x1   = dtb  + BL * DI_;               // (BL, D)      - after mamba residual
  float* ffnh = x1   + BL * D_;                // (BL, 2*D)    - ffn hidden
  float* cbuf = ffnh + BL * 2 * D_;            // (B, D)       - cls hidden

  float* outv = (float*)d_out;                 // (B,) logits
  float* x2   = outv + B_;                     // (BL, D) second tuple output

  // 1) LN1
  layernorm_kernel<<<dim3(M), dim3(256), 0, stream>>>(x, norm1_w, norm1_b, hln);

  // 2) in_proj: (BL,1024) x (4096,1024)^T -> xz
  launch_gemm(hln, D_, in_proj_W, D_, nullptr, nullptr, xz, 2 * DI_,
              M, 2 * DI_, D_, 0, stream);

  // 3) causal depthwise conv + SiLU -> uc
  {
    const int n = B_ * L_ * DI_;
    conv_silu_kernel<<<dim3((n + 255) / 256), dim3(256), 0, stream>>>(
        xz, conv_w, conv_b, uc);
  }

  // 4) x_proj: (BL,2048) x (96,2048)^T -> xdbl   (N=96 -> 32-wide tiles)
  launch_gemm(uc, DI_, x_proj_W, DI_, nullptr, nullptr, xdbl, XD_,
              M, XD_, DI_, 0, stream);

  // 5) dt_proj + bias + softplus: (BL,64 of xdbl) x (2048,64)^T -> dtb
  launch_gemm(xdbl, XD_, dt_proj_W, DTR_, dt_proj_b, nullptr, dtb, DI_,
              M, DI_, DTR_, 2, stream);

  // 6) SSM scan + D*u skip + SiLU(z) gate (in-place over uc)
  {
    const int n = B_ * DI_;
    ssm_scan_kernel<<<dim3((n + 255) / 256), dim3(256), 0, stream>>>(
        dtb, xdbl, xz, A_log, D_ssm, uc);
  }

  // 7) out_proj + residual(x): (BL,2048) x (1024,2048)^T -> x1
  launch_gemm(uc, DI_, out_projW, DI_, nullptr, x, x1, D_,
              M, D_, DI_, 0, stream);

  // 8) LN2
  layernorm_kernel<<<dim3(M), dim3(256), 0, stream>>>(x1, norm2_w, norm2_b, hln);

  // 9) FFN W1 + bias + SiLU
  launch_gemm(hln, D_, ffn_W1, D_, ffn_b1, nullptr, ffnh, 2 * D_,
              M, 2 * D_, D_, 1, stream);

  // 10) FFN W2 + bias + residual(x1) -> x2 (tuple output #2)
  launch_gemm(ffnh, 2 * D_, ffn_W2, 2 * D_, ffn_b2, x1, x2, D_,
              M, D_, 2 * D_, 0, stream);

  // 11) classifier stage 1: (B, L*D) x (1024, L*D)^T + bias + SiLU -> cbuf
  launch_gemm(x2, L_ * D_, cls_W1, L_ * D_, cls_b1, nullptr, cbuf, D_,
              B_, D_, L_ * D_, 1, stream);

  // 12) classifier stage 2: dot with cls_W2 + cls_b2 -> outv (tuple output #1)
  cls_head_kernel<<<dim3(B_), dim3(256), 0, stream>>>(cbuf, cls_W2, cls_b2, outv);
}
